// Custom_GraphConvolution_61314953118453
// MI455X (gfx1250) — matrix operand order repaired and checked by reading, not compiled, not run on hardware
//
#include <hip/hip_runtime.h>

typedef __attribute__((ext_vector_type(8)))  __bf16 v8bf;
typedef __attribute__((ext_vector_type(16))) __bf16 v16bf;
typedef __attribute__((ext_vector_type(8)))  float  v8f;

#define UDIM 128
#define FBOND 64
#define BN_EPS 1e-3f

// ---------------------------------------------------------------------------
// Fold BN into weights and pack into the WMMA B-operand lane layout (bf16).
// B tile (32x16, 16-bit): lane L<16 -> col N=L, holds K=0..7,16..23 (2 per VGPR);
// lanes 16..31 -> col N=L-16, K=8..15,24..31.  Packed so each lane reads 16
// consecutive bf16 (32 bytes) per (n-tile, k-step).
// packed[((nt*nk + kt)*32 + lane)*16 + j]
//   = W[(kt*32 + (j>>3)*16 + (lane>>4)*8 + (j&7))*128 + nt*16 + (lane&15)] * s
// ---------------------------------------------------------------------------
__global__ void pack_weights(const float* __restrict__ W,
                             const float* __restrict__ bvec,
                             const float* __restrict__ gamma,
                             const float* __restrict__ beta,
                             const float* __restrict__ mean,
                             const float* __restrict__ var,
                             __bf16* __restrict__ Wp,
                             float* __restrict__ biasp,
                             int fin) {
  int idx = blockIdx.x * blockDim.x + threadIdx.x;
  if (idx >= fin * UDIM) return;
  int k = idx / UDIM;
  int n = idx % UDIM;
  float s = gamma[n] * rsqrtf(var[n] + BN_EPS);
  float w = W[idx] * s;
  int kt   = k >> 5;
  int rem  = k & 31;
  int hi   = (rem >> 3) & 1;
  int j    = (rem & 7) | ((rem >> 4) << 3);
  int lane = hi * 16 + (n & 15);
  int nt   = n >> 4;
  int nk   = fin >> 5;
  Wp[(((nt * nk + kt) * 32 + lane) << 4) + j] = (__bf16)w;
  if (k == 0) biasp[n] = (bvec[n] - mean[n]) * s + beta[n];
}

// ---------------------------------------------------------------------------
// Kernel 1: atom_h = BN(atom_feat @ W1).  Writes fp32 into d_out (the
// "atom_h +" term) and a bf16 copy into workspace for the edge gather.
// Block = 256 threads = 8 waves; block owns 16 atoms; wave w owns cols
// [16w, 16w+16).
// ---------------------------------------------------------------------------
__global__ void __launch_bounds__(256)
atom_gemm(const float* __restrict__ atomf,
          const __bf16* __restrict__ W1p,
          const float* __restrict__ bias1,
          float* __restrict__ out,
          __bf16* __restrict__ atomh) {
  __shared__ __align__(16) __bf16 shA[16 * UDIM];   // 4 KB, bf16 rows

  const int tid = threadIdx.x;
  const size_t rowbase = (size_t)blockIdx.x * 16;

  // Stage 16 rows x 128 f32 -> bf16 LDS. 512 float4 loads over 256 threads.
  const float4* src = (const float4*)(atomf + rowbase * UDIM);
#pragma unroll
  for (int i = 0; i < 2; ++i) {
    float4 v = src[tid + i * 256];
    __bf16* p = shA + (size_t)(tid + i * 256) * 4;
    p[0] = (__bf16)v.x; p[1] = (__bf16)v.y; p[2] = (__bf16)v.z; p[3] = (__bf16)v.w;
  }
  __syncthreads();

  const int wave = tid >> 5;
  const int lane = tid & 31;
  const int m    = lane & 15;          // A row within tile
  const int kb   = (lane >> 4) * 8;    // K sub-block per A layout

  v8f acc = {};
#pragma unroll
  for (int kt = 0; kt < 4; ++kt) {     // K = 128 = 4 x 32
    v8bf lo = *(const v8bf*)(shA + m * UDIM + kt * 32 + kb);
    v8bf hi = *(const v8bf*)(shA + m * UDIM + kt * 32 + kb + 16);
    v16bf a = __builtin_shufflevector(lo, hi, 0,1,2,3,4,5,6,7,8,9,10,11,12,13,14,15);
    v16bf b = *(const v16bf*)(W1p + (((wave * 4 + kt) * 32 + lane) << 4));
    acc = __builtin_amdgcn_wmma_f32_16x16x32_bf16(false, a, false, b,
                                                  (short)0, acc, false, false);
  }

  const int col   = wave * 16 + (lane & 15);
  const int mhalf = (lane >> 4) * 8;
  const float bb  = bias1[col];
#pragma unroll
  for (int r = 0; r < 8; ++r) {        // C layout: M = r + 8*(lane>=16), N = lane&15
    size_t row = rowbase + (size_t)(mhalf + r);
    float v = acc[r] + bb;
    out[row * UDIM + col]   = v;
    atomh[row * UDIM + col] = (__bf16)v;
  }
}

// ---------------------------------------------------------------------------
// Kernel 2 (fused edge pipeline), 32 edges per block:
//   neigh = BN(atom_h[src] @ W2); bonds = BN(bond_feat @ W3);
//   atomicAdd(out[dst], neigh*bonds)
// The six B tiles (4 k-steps of W2 + 2 of W3) are loaded ONCE per wave into
// registers and reused across two 16-edge M-tiles, halving L2 weight traffic
// and block overhead.  Gather table (bf16 atom_h) and the scatter target both
// reside in the 192 MB L2.
// ---------------------------------------------------------------------------
__global__ void __launch_bounds__(256)
edge_gemm(const float* __restrict__ bondf,
          const int* __restrict__ pairs,
          const __bf16* __restrict__ atomh,
          const __bf16* __restrict__ W2p,
          const float* __restrict__ bias2,
          const __bf16* __restrict__ W3p,
          const float* __restrict__ bias3,
          float* __restrict__ out) {
  __shared__ __align__(16) __bf16 shN[32 * UDIM];   // gathered rows, 8 KB
  __shared__ __align__(16) __bf16 shB[32 * FBOND];  // bond rows, 4 KB
  __shared__ int shDst[32];
  __shared__ int shSrc[32];

  const int tid = threadIdx.x;
  const size_t ebase = (size_t)blockIdx.x * 32;

  // Stage edge indices once (avoids redundant per-lane pair loads).
  if (tid < 32) {
    shDst[tid] = pairs[(ebase + tid) * 2];
    shSrc[tid] = pairs[(ebase + tid) * 2 + 1];
  }
  __syncthreads();

  // Gather 32 bf16 atom_h rows: 32 rows x 16 chunks of 16 B = 512 chunks.
#pragma unroll
  for (int i = 0; i < 2; ++i) {
    int f  = tid + i * 256;
    int e  = f >> 4;
    int c8 = f & 15;
    ((v8bf*)(shN + e * UDIM))[c8] =
        ((const v8bf*)(atomh + (size_t)shSrc[e] * UDIM))[c8];
  }
  // Bond features f32 -> bf16: 32 x 64 = 2048 floats = 512 float4.
#pragma unroll
  for (int i = 0; i < 2; ++i) {
    int f = tid + i * 256;
    float4 v = ((const float4*)(bondf + ebase * FBOND))[f];
    __bf16* p = shB + f * 4;
    p[0] = (__bf16)v.x; p[1] = (__bf16)v.y; p[2] = (__bf16)v.z; p[3] = (__bf16)v.w;
  }
  __syncthreads();

  const int wave = tid >> 5;           // n-tile: cols [16*wave, 16*wave+16)
  const int lane = tid & 31;
  const int mr   = lane & 15;          // A row within M-tile
  const int kb   = (lane >> 4) * 8;    // K sub-block per A layout

  // Load the 6 B tiles once; reuse for both M-tiles.
  v16bf B2[4], B3[2];
#pragma unroll
  for (int kt = 0; kt < 4; ++kt)
    B2[kt] = *(const v16bf*)(W2p + (((wave * 4 + kt) * 32 + lane) << 4));
#pragma unroll
  for (int kt = 0; kt < 2; ++kt)
    B3[kt] = *(const v16bf*)(W3p + (((wave * 2 + kt) * 32 + lane) << 4));

  const int col   = wave * 16 + (lane & 15);
  const int mhalf = (lane >> 4) * 8;
  const float b2  = bias2[col];
  const float b3  = bias3[col];

#pragma unroll
  for (int mt = 0; mt < 2; ++mt) {
    const __bf16* rowN = shN + (mt * 16 + mr) * UDIM;
    const __bf16* rowB = shB + (mt * 16 + mr) * FBOND;

    v8f acc2 = {};                     // neigh @ W2, K = 128
#pragma unroll
    for (int kt = 0; kt < 4; ++kt) {
      v8bf lo = *(const v8bf*)(rowN + kt * 32 + kb);
      v8bf hi = *(const v8bf*)(rowN + kt * 32 + kb + 16);
      v16bf a = __builtin_shufflevector(lo, hi, 0,1,2,3,4,5,6,7,8,9,10,11,12,13,14,15);
      acc2 = __builtin_amdgcn_wmma_f32_16x16x32_bf16(false, a, false, B2[kt],
                                                     (short)0, acc2, false, false);
    }

    v8f acc3 = {};                     // bonds @ W3, K = 64
#pragma unroll
    for (int kt = 0; kt < 2; ++kt) {
      v8bf lo = *(const v8bf*)(rowB + kt * 32 + kb);
      v8bf hi = *(const v8bf*)(rowB + kt * 32 + kb + 16);
      v16bf a = __builtin_shufflevector(lo, hi, 0,1,2,3,4,5,6,7,8,9,10,11,12,13,14,15);
      acc3 = __builtin_amdgcn_wmma_f32_16x16x32_bf16(false, a, false, B3[kt],
                                                     (short)0, acc3, false, false);
    }

#pragma unroll
    for (int r = 0; r < 8; ++r) {      // C layout: M = r + 8*(lane>=16)
      float v = (acc2[r] + b2) * (acc3[r] + b3);
      atomicAdd(out + (size_t)shDst[mt * 16 + mhalf + r] * UDIM + col, v);
    }
  }
}

// ---------------------------------------------------------------------------
extern "C" void kernel_launch(void* const* d_in, const int* in_sizes, int n_in,
                              void* d_out, int out_size, void* d_ws, size_t ws_size,
                              hipStream_t stream) {
  const float* atomf = (const float*)d_in[0];
  const float* bondf = (const float*)d_in[1];
  const int*   pairs = (const int*)d_in[2];

  const float* W1 = (const float*)d_in[3];
  const float* b1 = (const float*)d_in[4];
  const float* g1 = (const float*)d_in[5];
  const float* be1= (const float*)d_in[6];
  const float* m1 = (const float*)d_in[7];
  const float* v1 = (const float*)d_in[8];
  const float* W2 = (const float*)d_in[9];
  const float* b2 = (const float*)d_in[10];
  const float* g2 = (const float*)d_in[11];
  const float* be2= (const float*)d_in[12];
  const float* m2 = (const float*)d_in[13];
  const float* v2 = (const float*)d_in[14];
  const float* W3 = (const float*)d_in[15];
  const float* b3 = (const float*)d_in[16];
  const float* g3 = (const float*)d_in[17];
  const float* be3= (const float*)d_in[18];
  const float* m3 = (const float*)d_in[19];
  const float* v3 = (const float*)d_in[20];

  const int N = in_sizes[0] / UDIM;    // 100000
  const int E = in_sizes[1] / FBOND;   // 800000

  // Workspace layout
  char* ws = (char*)d_ws;
  __bf16* W1p = (__bf16*)ws;  ws += (size_t)UDIM  * UDIM * sizeof(__bf16);
  __bf16* W2p = (__bf16*)ws;  ws += (size_t)UDIM  * UDIM * sizeof(__bf16);
  __bf16* W3p = (__bf16*)ws;  ws += (size_t)FBOND * UDIM * sizeof(__bf16);
  float* bias1 = (float*)ws;  ws += UDIM * sizeof(float);
  float* bias2 = (float*)ws;  ws += UDIM * sizeof(float);
  float* bias3 = (float*)ws;  ws += UDIM * sizeof(float);
  __bf16* atomh = (__bf16*)ws;            // N * 128 bf16 = 25.6 MB (L2-resident)

  float* out = (float*)d_out;

  pack_weights<<<(UDIM * UDIM + 255) / 256, 256, 0, stream>>>(
      W1, b1, g1, be1, m1, v1, W1p, bias1, UDIM);
  pack_weights<<<(UDIM * UDIM + 255) / 256, 256, 0, stream>>>(
      W2, b2, g2, be2, m2, v2, W2p, bias2, UDIM);
  pack_weights<<<(FBOND * UDIM + 255) / 256, 256, 0, stream>>>(
      W3, b3, g3, be3, m3, v3, W3p, bias3, FBOND);

  atom_gemm<<<N / 16, 256, 0, stream>>>(atomf, W1p, bias1, out, atomh);

  edge_gemm<<<E / 32, 256, 0, stream>>>(bondf, pairs, atomh,
                                        W2p, bias2, W3p, bias3, out);
}
